// GeometricAttention_67534065762993
// MI455X (gfx1250) — matrix operand order repaired and checked by reading, not compiled, not run on hardware
//
#include <hip/hip_runtime.h>
#include <hip/hip_bf16.h>

// ---------------- CDNA5 vector types for WMMA ----------------
typedef float  v8f   __attribute__((ext_vector_type(8)));
typedef float  v2f   __attribute__((ext_vector_type(2)));
typedef __bf16 v16bf __attribute__((ext_vector_type(16)));

#define B_  2
#define T_  2048
#define C_  1024
#define H_  16
#define BT_ (B_*T_)
#define HD_ 64
#define N3_ (H_*3)   // 48

// ---------------- wave32 helpers ----------------
static __device__ __forceinline__ int lane_id() {
  return (int)__builtin_amdgcn_mbcnt_lo(~0u, 0u); // wave32: lane id
}
static __device__ __forceinline__ float xor_bcast(float v, int mask) {
  int src = (lane_id() ^ mask) << 2;
  return __builtin_bit_cast(float, __builtin_amdgcn_ds_bpermute(src, __builtin_bit_cast(int, v)));
}

// ---------------- CDNA5 async global->LDS copy (ASYNCcnt-tracked) ----------------
// 16 bytes per active lane: LDS byte address in one VGPR, 64-bit global address in a pair.
static __device__ __forceinline__ void async_copy_b128(unsigned lds_addr, const void* gptr) {
  asm volatile("global_load_async_to_lds_b128 %0, %1, off"
               :: "v"(lds_addr), "v"(gptr)
               : "memory");
}
static __device__ __forceinline__ void wait_async0() {
  asm volatile("s_wait_asynccnt 0x0" ::: "memory");
}
// Low 32 bits of a generic pointer to __shared__ data are the LDS byte address.
static __device__ __forceinline__ unsigned lds_addr_of(const void* p) {
  return (unsigned)(unsigned long long)p;
}

// ---------------- fp32 -> bf16 conversion ----------------
__global__ void cvt_f32_bf16_kernel(const float* __restrict__ in,
                                    __bf16* __restrict__ out, int n) {
  int i = blockIdx.x * 256 + threadIdx.x;
  if (i < n) out[i] = (__bf16)in[i];
}

// ---------------- generic bf16 WMMA GEMM: C[M,N] = A[M,K] @ B[K,N] ----------------
// Block = 4 waves covering 64(M) x 64(N); each wave owns 16(M) x 64(N).
// B tile (32K x 64N) staged in LDS via async-to-LDS once per k-step (shared by all waves).
template <bool OUT_BF16>
__global__ __launch_bounds__(128)
void gemm_bf16_kernel(const __bf16* __restrict__ A, const __bf16* __restrict__ Bm,
                      void* __restrict__ Cout, int M, int N, int K) {
  __shared__ __bf16 sB[32][72];                  // 32 k-rows x 64 n-cols (padded, 16B-aligned rows)
  const int tid  = threadIdx.x;
  const int lane = tid & 31;
  const int wid  = tid >> 5;
  const int ml   = lane & 15;
  const int hi   = lane >> 4;                    // half-wave select
  const int m0   = blockIdx.x * 64 + wid * 16;
  const int n0   = blockIdx.y * 64;
  const int row  = m0 + ml;
  const unsigned sBbase = lds_addr_of(&sB[0][0]);

  v8f acc[4];
#pragma unroll
  for (int nb = 0; nb < 4; ++nb)
#pragma unroll
    for (int r = 0; r < 8; ++r) acc[nb][r] = 0.0f;

  for (int kb = 0; kb < K; kb += 32) {
    __syncthreads();                             // sB reuse fence
    // ---- async stage of B tile: 256 chunks of 16B, 2 per thread ----
    for (int c = tid; c < 256; c += 128) {
      const int r   = c >> 3;                    // k-row within tile, 0..31
      const int col = (c & 7) * 8;               // n-col start, 0..56
      if (n0 + col < N) {                        // N is a multiple of 16 -> chunk fully in range
        async_copy_b128(sBbase + (unsigned)(r * 72 + col) * 2,
                        Bm + (size_t)(kb + r) * N + n0 + col);
      }
    }
    // prefetch next k-step while the async copies are in flight
    if (kb + 32 < K) {
      __builtin_prefetch(A + (size_t)row * K + kb + 32, 0, 0);
      __builtin_prefetch(Bm + (size_t)(kb + 32 + (tid & 31)) * N + n0, 0, 0);
    }

    // ---- A fragment 16x32 bf16 (lo lanes K 0-7 & 16-23, hi lanes K 8-15 & 24-31) ----
    v16bf afrag;
    const __bf16* ap = A + (size_t)row * K + kb + hi * 8;
#pragma unroll
    for (int j = 0; j < 8; ++j) afrag[j]     = ap[j];
#pragma unroll
    for (int j = 0; j < 8; ++j) afrag[8 + j] = ap[16 + j];

    wait_async0();
    __syncthreads();

#pragma unroll
    for (int nb = 0; nb < 4; ++nb) {
      if (n0 + nb * 16 >= N) continue;           // uniform across wave
      v16bf bfrag;                               // B 32x16: lo lanes K 0-7 & 16-23, hi lanes K 8-15 & 24-31
#pragma unroll
      for (int j = 0; j < 16; ++j) bfrag[j] = sB[((j & 8) << 1) + hi * 8 + (j & 7)][nb * 16 + ml];
      acc[nb] = __builtin_amdgcn_wmma_f32_16x16x32_bf16(
          false, afrag, false, bfrag, (short)0, acc[nb], false, false);
    }
  }
#pragma unroll
  for (int nb = 0; nb < 4; ++nb) {
    if (n0 + nb * 16 >= N) continue;
    const int n = n0 + nb * 16 + ml;
#pragma unroll
    for (int r = 0; r < 8; ++r) {
      const int m = m0 + r + hi * 8;             // C layout: VGPR r -> row r (+8 for hi lanes)
      if (OUT_BF16) ((__bf16*)Cout)[(size_t)m * N + n] = (__bf16)acc[nb][r];
      else          ((float*) Cout)[(size_t)m * N + n] = acc[nb][r];
    }
  }
}

// ---------------- fold geo bias into 4-dim q/k ----------------
__global__ void build_qeke_kernel(const float* __restrict__ q3, const float* __restrict__ k3,
                                  const float* __restrict__ dirs, const float* __restrict__ scale,
                                  float* __restrict__ qe, float* __restrict__ ke) {
  int idx = blockIdx.x * 256 + threadIdx.x;      // over B*T*H
  if (idx >= BT_ * H_) return;
  int h = idx % H_;
  int bt = idx / H_;
  int t = bt % T_;
  int b = bt / T_;
  const float* q = q3 + (size_t)bt * N3_ + h * 3;
  const float* k = k3 + (size_t)bt * N3_ + h * 3;
  float d0 = dirs[h * 3], d1 = dirs[h * 3 + 1], d2 = dirs[h * 3 + 2];
  float sc = scale[0];
  float qon = q[0] * d0 + q[1] * d1 + q[2] * d2;
  float kon = k[0] * d0 + k[1] * d1 + k[2] * d2;
  size_t o = (((size_t)b * H_ + h) * T_ + t) * 4;
  qe[o] = q[0] * sc; qe[o + 1] = q[1] * sc; qe[o + 2] = q[2] * sc; qe[o + 3] = qon;
  ke[o] = k[0];      ke[o + 1] = k[1];      ke[o + 2] = k[2];      ke[o + 3] = kon;
}

// ---------------- flash attention ----------------
// grid: (T/64, H, B), block: 128 (4 waves, 16 queries each).
// scores: v_wmma_f32_16x16x4_f32 (fp32, 4-dim qe/ke); P@V: v_wmma_f32_16x16x32_bf16.
// V tiles staged in LDS via async-to-LDS.
__global__ __launch_bounds__(128)
void attn_kernel(const float* __restrict__ qe, const float* __restrict__ ke,
                 const __bf16* __restrict__ vb, __bf16* __restrict__ ao) {
  const int lane = threadIdx.x & 31;
  const int wid  = threadIdx.x >> 5;
  const int ml   = lane & 15;
  const int hi   = lane >> 4;
  const int h = blockIdx.y, b = blockIdx.z;
  const int blockStart = blockIdx.x * 64;
  const int q0 = blockStart + wid * 16;
  const float NEG_INF = -__builtin_inff();

  __shared__ __bf16 sV[32][72];        // 32 keys x 64 dims (padded, 16B-aligned rows)
  __shared__ __bf16 sP[4][16][34];     // per-wave P tile 16x32 (padded)
  const unsigned sVbase = lds_addr_of(&sV[0][0]);

  const float* qeb = qe + (((size_t)b * H_ + h) * T_) * 4;
  const float* keb = ke + (((size_t)b * H_ + h) * T_) * 4;

  // qe A-fragment (32-bit A 16x4): lo lanes K={0,1}, hi lanes K={2,3}
  v2f afq;
  afq[0] = qeb[(size_t)(q0 + ml) * 4 + hi * 2 + 0];
  afq[1] = qeb[(size_t)(q0 + ml) * 4 + hi * 2 + 1];

  float mrun[8], lrun[8];
  v8f acc[4];
#pragma unroll
  for (int r = 0; r < 8; ++r) { mrun[r] = -3.0e38f; lrun[r] = 0.0f; }
#pragma unroll
  for (int nb = 0; nb < 4; ++nb)
#pragma unroll
    for (int r = 0; r < 8; ++r) acc[nb][r] = 0.0f;

  const int nTiles = blockStart / 32 + 2;        // uniform over the workgroup
  for (int tI = 0; tI < nTiles; ++tI) {
    const int s0 = tI * 32;
    __syncthreads();                             // protect sV before overwrite
    // ---- async stage of V tile: 32 keys x 64 dims, 16B chunks ----
    for (int c = threadIdx.x; c < 256; c += 128) {
      const int r   = c >> 3;                    // key row within tile
      const int col = (c & 7) * 8;               // dim start
      async_copy_b128(sVbase + (unsigned)(r * 72 + col) * 2,
                      vb + ((size_t)b * T_ + s0 + r) * C_ + h * HD_ + col);
    }

    // ---- ke B-fragments for both 16-key halves (issue both loads up front) ----
    v2f bfk0, bfk1;                              // 32-bit B 4x16: lo lanes K={0,1}, hi K={2,3}
    bfk0[0] = keb[(size_t)(s0 + ml) * 4 + hi * 2 + 0];
    bfk0[1] = keb[(size_t)(s0 + ml) * 4 + hi * 2 + 1];
    bfk1[0] = keb[(size_t)(s0 + 16 + ml) * 4 + hi * 2 + 0];
    bfk1[1] = keb[(size_t)(s0 + 16 + ml) * 4 + hi * 2 + 1];

    v8f z;
#pragma unroll
    for (int r = 0; r < 8; ++r) z[r] = 0.0f;
    v8f sc[2];
    sc[0] = __builtin_amdgcn_wmma_f32_16x16x4_f32(false, afq, false, bfk0, (short)0, z, false, false);
    sc[1] = __builtin_amdgcn_wmma_f32_16x16x4_f32(false, afq, false, bfk1, (short)0, z, false, false);

    // causal mask + online softmax (row = r + hi*8, cols spread over 16 lanes)
    float p0s[8], p1s[8];
#pragma unroll
    for (int r = 0; r < 8; ++r) {
      const int q = q0 + r + hi * 8;
      float s0v = sc[0][r], s1v = sc[1][r];
      if (s0 + ml      > q) s0v = NEG_INF;
      if (s0 + 16 + ml > q) s1v = NEG_INF;
      float mt = fmaxf(s0v, s1v);
      mt = fmaxf(mt, xor_bcast(mt, 1));
      mt = fmaxf(mt, xor_bcast(mt, 2));
      mt = fmaxf(mt, xor_bcast(mt, 4));
      mt = fmaxf(mt, xor_bcast(mt, 8));
      const float mn = fmaxf(mrun[r], mt);
      const float p0 = __expf(s0v - mn);
      const float p1 = __expf(s1v - mn);
      float ls = p0 + p1;
      ls += xor_bcast(ls, 1);
      ls += xor_bcast(ls, 2);
      ls += xor_bcast(ls, 4);
      ls += xor_bcast(ls, 8);
      const float corr = __expf(mrun[r] - mn);
      lrun[r] = lrun[r] * corr + ls;
      mrun[r] = mn;
#pragma unroll
      for (int nb = 0; nb < 4; ++nb) acc[nb][r] *= corr;
      p0s[r] = p0; p1s[r] = p1;
    }

    // repack P (C layout) -> bf16 A fragment layout via LDS
#pragma unroll
    for (int r = 0; r < 8; ++r) {
      sP[wid][r + hi * 8][ml]      = (__bf16)p0s[r];
      sP[wid][r + hi * 8][16 + ml] = (__bf16)p1s[r];
    }
    wait_async0();                               // V tile resident
    __syncthreads();                             // P + V visible

    v16bf pa;
#pragma unroll
    for (int j = 0; j < 8; ++j) pa[j]     = sP[wid][ml][hi * 8 + j];
#pragma unroll
    for (int j = 0; j < 8; ++j) pa[8 + j] = sP[wid][ml][16 + hi * 8 + j];

#pragma unroll
    for (int nb = 0; nb < 4; ++nb) {
      v16bf vf;                                  // B 32x16: lo lanes K 0-7 & 16-23, hi lanes K 8-15 & 24-31
#pragma unroll
      for (int j = 0; j < 16; ++j) vf[j] = sV[((j & 8) << 1) + hi * 8 + (j & 7)][nb * 16 + ml];
      acc[nb] = __builtin_amdgcn_wmma_f32_16x16x32_bf16(
          false, pa, false, vf, (short)0, acc[nb], false, false);
    }
  }

  // normalize and store in (B,T,H*64) layout (matches reference reshape)
#pragma unroll
  for (int nb = 0; nb < 4; ++nb) {
    const int d = nb * 16 + ml;
#pragma unroll
    for (int r = 0; r < 8; ++r) {
      const int q = q0 + r + hi * 8;
      ao[((size_t)b * T_ + q) * C_ + h * HD_ + d] = (__bf16)(acc[nb][r] / lrun[r]);
    }
  }
}

// ---------------- launcher ----------------
extern "C" void kernel_launch(void* const* d_in, const int* in_sizes, int n_in,
                              void* d_out, int out_size, void* d_ws, size_t ws_size,
                              hipStream_t stream) {
  const float* x    = (const float*)d_in[0];
  const float* wq   = (const float*)d_in[1];
  const float* wk   = (const float*)d_in[2];
  const float* wv   = (const float*)d_in[3];
  const float* wo   = (const float*)d_in[4];
  const float* dirs = (const float*)d_in[5];
  const float* scal = (const float*)d_in[6];

  char* ws = (char*)d_ws;
  size_t off = 0;
  auto alloc = [&](size_t bytes) { size_t o = off; off += (bytes + 255) & ~(size_t)255; return o; };
  const size_t o_xb  = alloc((size_t)BT_ * C_ * 2);
  const size_t o_wqb = alloc((size_t)C_ * N3_ * 2);
  const size_t o_wkb = alloc((size_t)C_ * N3_ * 2);
  const size_t o_wvb = alloc((size_t)C_ * C_ * 2);
  const size_t o_wob = alloc((size_t)C_ * C_ * 2);
  const size_t o_vb  = alloc((size_t)BT_ * C_ * 2);
  const size_t o_q3  = alloc((size_t)BT_ * N3_ * 4);
  const size_t o_k3  = alloc((size_t)BT_ * N3_ * 4);
  const size_t o_qe  = alloc((size_t)B_ * H_ * T_ * 4 * 4);
  const size_t o_ke  = alloc((size_t)B_ * H_ * T_ * 4 * 4);
  const size_t o_ao  = alloc((size_t)BT_ * C_ * 2);
  if (ws_size < off) return;

  __bf16* xb  = (__bf16*)(ws + o_xb);
  __bf16* wqb = (__bf16*)(ws + o_wqb);
  __bf16* wkb = (__bf16*)(ws + o_wkb);
  __bf16* wvb = (__bf16*)(ws + o_wvb);
  __bf16* wob = (__bf16*)(ws + o_wob);
  __bf16* vb  = (__bf16*)(ws + o_vb);
  float*  q3  = (float*)(ws + o_q3);
  float*  k3  = (float*)(ws + o_k3);
  float*  qe  = (float*)(ws + o_qe);
  float*  ke  = (float*)(ws + o_ke);
  __bf16* ao  = (__bf16*)(ws + o_ao);

  auto cvt = [&](const float* src, __bf16* dst, int n) {
    cvt_f32_bf16_kernel<<<(n + 255) / 256, 256, 0, stream>>>(src, dst, n);
  };
  cvt(x,  xb,  BT_ * C_);
  cvt(wq, wqb, C_ * N3_);
  cvt(wk, wkb, C_ * N3_);
  cvt(wv, wvb, C_ * C_);
  cvt(wo, wob, C_ * C_);

  // q3/k3 projections (fp32 out), v projection (bf16 out)
  gemm_bf16_kernel<false><<<dim3(BT_ / 64, 1), 128, 0, stream>>>(xb, wqb, q3, BT_, N3_, C_);
  gemm_bf16_kernel<false><<<dim3(BT_ / 64, 1), 128, 0, stream>>>(xb, wkb, k3, BT_, N3_, C_);
  gemm_bf16_kernel<true ><<<dim3(BT_ / 64, C_ / 64), 128, 0, stream>>>(xb, wvb, vb, BT_, C_, C_);

  build_qeke_kernel<<<(BT_ * H_ + 255) / 256, 256, 0, stream>>>(q3, k3, dirs, scal, qe, ke);

  attn_kernel<<<dim3(T_ / 64, H_, B_), 128, 0, stream>>>(qe, ke, vb, ao);

  // final projection straight into d_out (fp32)
  gemm_bf16_kernel<false><<<dim3(BT_ / 64, C_ / 64), 128, 0, stream>>>(ao, wob, (float*)d_out, BT_, C_, C_);
}
